// PixelwiseBoundaryContrastiveLoss_88038239634237
// MI455X (gfx1250) — matrix operand order repaired and checked by reading, not compiled, not run on hardware
//
#include <hip/hip_runtime.h>

// ---------------- problem constants (from reference) ----------------
#define BATCH   2
#define CHAN    32
#define DDIM    64
#define HDIM    128
#define WDIM    128
#define NVOX    (DDIM * HDIM * WDIM)      // 1,048,576 (power of two)
#define P_CAP   2048
#define NEG_CAP 12288
#define KPAD    5                          // (11-1)/2
#define TAU_INV 10.0f
#define EPSL    1e-8f
#define FNEG    -3.0e38f
#define FNEG_THRESH -1.0e30f

#define CHUNK    4096
#define NCHUNK   (NVOX / CHUNK)            // 256
#define TPOS_TILES (P_CAP / 16)            // 128
#define TNEG_TILES (NEG_CAP / 16)          // 768
#define TTOT_TILES (TPOS_TILES + TNEG_TILES)

typedef __attribute__((ext_vector_type(16))) _Float16 v16h;
typedef __attribute__((ext_vector_type(8)))  float    v8f;
typedef __attribute__((ext_vector_type(4)))  int      i32x4;

union VU { v16h v; i32x4 q[2]; };

// ---------------- workspace layout (bytes, 256-aligned) ----------------
#define OFF_FG      ((size_t)0)
#define OFF_TA      ((size_t)(OFF_FG + (size_t)BATCH * NVOX))
#define OFF_TB      ((size_t)(OFF_TA + (size_t)BATCH * NVOX))
#define OFF_CNTFG   ((size_t)(OFF_TB + (size_t)BATCH * NVOX))
#define OFF_CNTRIM  ((size_t)(OFF_CNTFG  + BATCH * NCHUNK * 4))
#define OFF_OFFFG   ((size_t)(OFF_CNTRIM + BATCH * NCHUNK * 4))
#define OFF_OFFRIM  ((size_t)(OFF_OFFFG  + BATCH * NCHUNK * 4))
#define OFF_HDR     ((size_t)(OFF_OFFRIM + BATCH * NCHUNK * 4))
#define OFF_POSIDX  ((size_t)(OFF_HDR + 256))
#define OFF_NEGIDX  ((size_t)(OFF_POSIDX + (size_t)BATCH * P_CAP * 4))
#define OFF_POSF    ((size_t)(OFF_NEGIDX + (size_t)BATCH * NEG_CAP * 4))
#define OFF_NEGF    ((size_t)(OFF_POSF + (size_t)BATCH * P_CAP * CHAN * 2))
#define OFF_LOSSP   ((size_t)(OFF_NEGF + (size_t)BATCH * NEG_CAP * CHAN * 2))

// ---------------- kernels ----------------

__global__ __launch_bounds__(256) void fg_kernel(const int* __restrict__ labels,
                                                 unsigned char* __restrict__ fg, int total) {
  int i = blockIdx.x * 256 + threadIdx.x;
  if (i < total) fg[i] = labels[i] > 0 ? 1 : 0;
}

// Separable binary dilation (box max), AXIS: 0=W(stride1,len128) 1=H(stride128,len128) 2=D(stride16384,len64)
template <int AXIS>
__global__ __launch_bounds__(256) void dil_kernel(const unsigned char* __restrict__ in,
                                                  unsigned char* __restrict__ out, int total) {
  int i = blockIdx.x * 256 + threadIdx.x;
  if (i >= total) return;
  int n = i & (NVOX - 1);
  int coord, stride, len;
  if (AXIS == 0)      { coord = n & (WDIM - 1);          stride = 1;            len = WDIM; }
  else if (AXIS == 1) { coord = (n >> 7) & (HDIM - 1);   stride = WDIM;         len = HDIM; }
  else                { coord = n >> 14;                 stride = HDIM * WDIM;  len = DDIM; }
  int lo = (coord - KPAD < 0 ? 0 : coord - KPAD) - coord;
  int hi = (coord + KPAD > len - 1 ? len - 1 : coord + KPAD) - coord;
  unsigned char acc = 0;
  for (int d = lo; d <= hi; ++d) acc |= in[i + d * stride];
  out[i] = acc;
}

__global__ __launch_bounds__(256) void count_kernel(const unsigned char* __restrict__ fg,
                                                    const unsigned char* __restrict__ dil,
                                                    int* __restrict__ cntFg, int* __restrict__ cntRim) {
  int chunk = blockIdx.x, b = blockIdx.y, t = threadIdx.x;
  size_t base = (size_t)b * NVOX + (size_t)chunk * CHUNK + (size_t)t * 16;
  int cf = 0, cr = 0;
  for (int i = 0; i < 16; ++i) {
    unsigned char f = fg[base + i], d = dil[base + i];
    cf += f; cr += (d && !f) ? 1 : 0;
  }
  __shared__ int sF[256], sR[256];
  sF[t] = cf; sR[t] = cr; __syncthreads();
  for (int s = 128; s > 0; s >>= 1) {
    if (t < s) { sF[t] += sF[t + s]; sR[t] += sR[t + s]; }
    __syncthreads();
  }
  if (t == 0) { cntFg[b * NCHUNK + chunk] = sF[0]; cntRim[b * NCHUNK + chunk] = sR[0]; }
}

__global__ void scan_kernel(const int* __restrict__ cntFg, const int* __restrict__ cntRim,
                            int* __restrict__ offFg, int* __restrict__ offRim, int* __restrict__ hdr) {
  int b = threadIdx.x;
  if (b >= BATCH || blockIdx.x != 0) return;
  int accF = 0, accR = 0;
  for (int i = 0; i < NCHUNK; ++i) {
    offFg[b * NCHUNK + i] = accF;  accF += cntFg[b * NCHUNK + i];
    offRim[b * NCHUNK + i] = accR; accR += cntRim[b * NCHUNK + i];
  }
  hdr[0 + b] = accF;                                // n_pos total (uncapped)
  hdr[2 + b] = accR;                                // n_rim total
  hdr[4 + b] = accF < P_CAP   ? accF : P_CAP;       // selected pos count
  hdr[6 + b] = accR < NEG_CAP ? accR : NEG_CAP;     // selected neg count
}

__global__ __launch_bounds__(256) void select_kernel(const unsigned char* __restrict__ fg,
                                                     const unsigned char* __restrict__ dil,
                                                     const int* __restrict__ offFg,
                                                     const int* __restrict__ offRim,
                                                     int* __restrict__ posIdx, int* __restrict__ negIdx) {
  int chunk = blockIdx.x, b = blockIdx.y, t = threadIdx.x;
  size_t base = (size_t)b * NVOX + (size_t)chunk * CHUNK + (size_t)t * 16;
  int n0 = chunk * CHUNK + t * 16;
  unsigned char fv[16], rv[16];
  int cf = 0, cr = 0;
  for (int i = 0; i < 16; ++i) {
    unsigned char f = fg[base + i], d = dil[base + i];
    fv[i] = f; rv[i] = (d && !f) ? 1 : 0;
    cf += fv[i]; cr += rv[i];
  }
  __shared__ int sF[256], sR[256];
  sF[t] = cf; sR[t] = cr; __syncthreads();
  if (t == 0) {  // small serial exclusive scan (256 elems) - deterministic, cheap
    int rF = 0, rR = 0;
    for (int i = 0; i < 256; ++i) {
      int a = sF[i]; sF[i] = rF; rF += a;
      int c = sR[i]; sR[i] = rR; rR += c;
    }
  }
  __syncthreads();
  int rF = offFg[b * NCHUNK + chunk] + sF[t];
  int rR = offRim[b * NCHUNK + chunk] + sR[t];
  for (int i = 0; i < 16; ++i) {
    if (fv[i]) { if (rF < P_CAP)   posIdx[b * P_CAP + rF]   = n0 + i; rF++; }
    if (rv[i]) { if (rR < NEG_CAP) negIdx[b * NEG_CAP + rR] = n0 + i; rR++; }
  }
}

// One wave32 per selected voxel; lane == channel. L2-normalize, store fp16 row of 32.
__global__ __launch_bounds__(256) void gather_kernel(const float* __restrict__ feat,
                                                     const int* __restrict__ hdr,
                                                     const int* __restrict__ posIdx,
                                                     const int* __restrict__ negIdx,
                                                     _Float16* __restrict__ posF,
                                                     _Float16* __restrict__ negF) {
  int gw = blockIdx.x * 8 + (threadIdx.x >> 5);
  int lane = threadIdx.x & 31;
  int b = gw / (P_CAP + NEG_CAP);
  int s = gw % (P_CAP + NEG_CAP);
  if (b >= BATCH) return;
  bool isPos = s < P_CAP;
  int slot = isPos ? s : s - P_CAP;
  int count = isPos ? hdr[4 + b] : hdr[6 + b];
  _Float16* row = isPos ? (posF + ((size_t)b * P_CAP + slot) * CHAN)
                        : (negF + ((size_t)b * NEG_CAP + slot) * CHAN);
  float out = 0.0f;
  if (slot < count) {
    int idx = isPos ? posIdx[b * P_CAP + slot] : negIdx[b * NEG_CAP + slot];
    float v = feat[((size_t)b * CHAN + lane) * NVOX + idx];
    float ss = v * v;
    #pragma unroll
    for (int mk = 1; mk < 32; mk <<= 1) ss += __shfl_xor(ss, mk, 32);
    out = v / fmaxf(sqrtf(ss), 1e-12f);
  }
  row[lane] = (_Float16)out;   // zero-pad unused slots (masked out anyway)
}

// Main contrastive-loss kernel.
// Tile mapping: D = neighbors(16x32, A operand) x anchors^T(32x16, B operand).
// In the f32 C/D layout, N (= anchor) is the lane index and M (= neighbor)
// spans the 8 accumulator VGPRs, so the per-anchor online softmax over
// neighbors is a pure in-lane reduction -> no cross-lane ops in the loop.
__global__ __launch_bounds__(256) void loss_kernel(const _Float16* __restrict__ posF,
                                                   const _Float16* __restrict__ negF,
                                                   const int* __restrict__ hdr,
                                                   float* __restrict__ lossPartial) {
  const int mTile = blockIdx.x;          // 0..127 anchor tiles
  const int b = blockIdx.y;
  const int nPos = hdr[4 + b];
  const int nNeg = hdr[6 + b];
  const int rowBase = mTile * 16;        // first anchor slot of this block

  __shared__ float sM[8][16], sA[8][16], sP[8][16];
  __shared__ float sRowLoss[16];

  if (rowBase >= nPos || nPos <= 0) {
    if (threadIdx.x == 0) lossPartial[b * TPOS_TILES + mTile] = 0.0f;
    return;
  }

  const int tid  = threadIdx.x;
  const int wave = tid >> 5;
  const int lane = tid & 31;
  const int half = lane >> 4;     // 0 or 1
  const int nIdx = lane & 15;     // this lane's anchor within the tile

  const _Float16* posBase = posF + (size_t)b * P_CAP * CHAN;
  const _Float16* negBase = negF + (size_t)b * NEG_CAP * CHAN;

  // B operand (anchors, fixed): contiguous K order (K = 16*half + s), same as A:
  // lane<16: K 0..15, N=nIdx; lane>=16: K 16..31, N=nIdx
  VU bm;
  {
    const _Float16* cb = posBase + (size_t)(rowBase + nIdx) * CHAN;
    bm.q[0] = *(const i32x4*)(cb + half * 16);
    bm.q[1] = *(const i32x4*)(cb + half * 16 + 8);
  }

  const int anchorSlot = rowBase + nIdx;

  // Per-lane online-softmax state for this lane's anchor (half-partial).
  float m = FNEG, sAll = 0.0f, sPos = 0.0f;

  for (int t = wave; t < TTOT_TILES; t += 8) {
    const bool posRegion = t < TPOS_TILES;
    const int colBase = posRegion ? t * 16 : (t - TPOS_TILES) * 16;
    const int cnt = posRegion ? nPos : nNeg;
    if (colBase >= cnt) continue;                       // wave-uniform skip

    // A operand (neighbor rows), K order matching B:
    // lane<16 holds row nIdx, K 0..15; lane>=16 holds K 16..31
    VU am;
    {
      const _Float16* rp = (posRegion ? posBase : negBase) + (size_t)(colBase + nIdx) * CHAN;
      const int o = half * 16;  // in halfs
      am.q[0] = *(const i32x4*)(rp + o);
      am.q[1] = *(const i32x4*)(rp + o + 8);
    }
    v8f c = {};
    c = __builtin_amdgcn_wmma_f32_16x16x32_f16(false, am.v, false, bm.v, (short)0, c, false, false);

    // Masked logits for the 8 neighbors this lane sees (all in-lane).
    float lv[8];
    #pragma unroll
    for (int r = 0; r < 8; ++r) {
      const int neighSlot = colBase + r + half * 8;
      const bool valid = (neighSlot < cnt) && (!posRegion || neighSlot != anchorSlot);
      lv[r] = valid ? c[r] * TAU_INV : FNEG;
    }
    float vmax = lv[0];
    #pragma unroll
    for (int r = 1; r < 8; ++r) vmax = fmaxf(vmax, lv[r]);
    const float newm  = fmaxf(m, vmax);
    const float scale = __expf(m - newm);
    float esum = 0.0f;
    #pragma unroll
    for (int r = 0; r < 8; ++r) {
      const float e = (lv[r] > FNEG_THRESH) ? __expf(lv[r] - newm) : 0.0f;
      esum += e;
    }
    sAll = sAll * scale + esum;
    sPos = posRegion ? (sPos * scale + esum) : (sPos * scale);
    m = newm;
  }

  // Merge the two 16-lane halves (same anchor, disjoint neighbor sets).
  {
    const float mo = __shfl_xor(m, 16, 32);
    const float ao = __shfl_xor(sAll, 16, 32);
    const float po = __shfl_xor(sPos, 16, 32);
    const float mm = fmaxf(m, mo);
    const float s0 = __expf(m - mm), s1 = __expf(mo - mm);
    sAll = sAll * s0 + ao * s1;
    sPos = sPos * s0 + po * s1;
    m = mm;
  }

  if (half == 0) { sM[wave][nIdx] = m; sA[wave][nIdx] = sAll; sP[wave][nIdx] = sPos; }
  __syncthreads();

  if (tid < 16) {
    float mm = FNEG;
    for (int w = 0; w < 8; ++w) mm = fmaxf(mm, sM[w][tid]);
    float aSum = 0.0f, pSum = 0.0f;
    for (int w = 0; w < 8; ++w) {
      const float sc = __expf(sM[w][tid] - mm);
      aSum += sA[w][tid] * sc;
      pSum += sP[w][tid] * sc;
    }
    const int gi = rowBase + tid;
    sRowLoss[tid] = (gi < nPos) ? -(__logf(pSum + EPSL) - __logf(aSum + EPSL)) : 0.0f;
  }
  __syncthreads();

  if (tid == 0) {
    float s = 0.0f;
    for (int i = 0; i < 16; ++i) s += sRowLoss[i];     // fixed order -> deterministic
    lossPartial[b * TPOS_TILES + mTile] = s;
  }
}

__global__ void final_kernel(const float* __restrict__ lossPartial,
                             const int* __restrict__ hdr, float* __restrict__ out) {
  if (threadIdx.x != 0 || blockIdx.x != 0) return;
  float tot = 0.0f; int nv = 0;
  for (int b = 0; b < BATCH; ++b) {
    float s = 0.0f;
    for (int k = 0; k < TPOS_TILES; ++k) s += lossPartial[b * TPOS_TILES + k];
    const int cnt = hdr[4 + b];
    const float imgLoss = s / (float)(cnt > 1 ? cnt : 1);
    const bool valid = (hdr[0 + b] >= 2) && (hdr[2 + b] > 0);
    if (valid) { tot += imgLoss; nv++; }
  }
  out[0] = (nv > 0) ? tot / (float)nv : 0.0f;
}

// ---------------- launch ----------------
extern "C" void kernel_launch(void* const* d_in, const int* in_sizes, int n_in,
                              void* d_out, int out_size, void* d_ws, size_t ws_size,
                              hipStream_t stream) {
  const float* features = (const float*)d_in[0];
  const int*   labels   = (const int*)d_in[1];
  float* out = (float*)d_out;
  char* ws = (char*)d_ws;

  unsigned char* fg = (unsigned char*)(ws + OFF_FG);
  unsigned char* tA = (unsigned char*)(ws + OFF_TA);
  unsigned char* tB = (unsigned char*)(ws + OFF_TB);
  int* cntFg  = (int*)(ws + OFF_CNTFG);
  int* cntRim = (int*)(ws + OFF_CNTRIM);
  int* offFg  = (int*)(ws + OFF_OFFFG);
  int* offRim = (int*)(ws + OFF_OFFRIM);
  int* hdr    = (int*)(ws + OFF_HDR);
  int* posIdx = (int*)(ws + OFF_POSIDX);
  int* negIdx = (int*)(ws + OFF_NEGIDX);
  _Float16* posF = (_Float16*)(ws + OFF_POSF);
  _Float16* negF = (_Float16*)(ws + OFF_NEGF);
  float* lossP = (float*)(ws + OFF_LOSSP);

  const int totalVox = BATCH * NVOX;
  const int nb = totalVox / 256;

  fg_kernel<<<nb, 256, 0, stream>>>(labels, fg, totalVox);
  dil_kernel<0><<<nb, 256, 0, stream>>>(fg, tA, totalVox);
  dil_kernel<1><<<nb, 256, 0, stream>>>(tA, tB, totalVox);
  dil_kernel<2><<<nb, 256, 0, stream>>>(tB, tA, totalVox);   // tA = dilated mask
  count_kernel<<<dim3(NCHUNK, BATCH), 256, 0, stream>>>(fg, tA, cntFg, cntRim);
  scan_kernel<<<1, 32, 0, stream>>>(cntFg, cntRim, offFg, offRim, hdr);
  select_kernel<<<dim3(NCHUNK, BATCH), 256, 0, stream>>>(fg, tA, offFg, offRim, posIdx, negIdx);
  gather_kernel<<<BATCH * (P_CAP + NEG_CAP) / 8, 256, 0, stream>>>(features, hdr, posIdx, negIdx, posF, negF);
  loss_kernel<<<dim3(TPOS_TILES, BATCH), 256, 0, stream>>>(posF, negF, hdr, lossP);
  final_kernel<<<1, 1, 0, stream>>>(lossP, hdr, out);
}
